// GraphPooling_10376640987639
// MI455X (gfx1250) — matrix operand order repaired and checked by reading, not compiled, not run on hardware
//
#include <hip/hip_runtime.h>
#include <hip/hip_bf16.h>

typedef __attribute__((ext_vector_type(16))) __bf16 v16bf;
typedef __attribute__((ext_vector_type(8)))  __bf16 v8bf;
typedef __attribute__((ext_vector_type(8)))  float  v8f;

#define NEG_ATT 0.2f
#define NEG_ACT 0.1f

// ---------- helpers ----------

__device__ __forceinline__ unsigned short f2bf(float f) {
  // round-to-nearest-even fp32 -> bf16
  unsigned int u = __float_as_uint(f);
  u += 0x7FFFu + ((u >> 16) & 1u);
  return (unsigned short)(u >> 16);
}

__device__ __forceinline__ void atomicMaxFloat(float* addr, float val) {
  unsigned int* ua = (unsigned int*)addr;
  unsigned int old = *ua;
  while (true) {
    float f = __uint_as_float(old);
    if (f >= val) break;
    unsigned int assumed = old;
    old = atomicCAS(ua, assumed, __float_as_uint(val));
    if (old == assumed) break;
  }
}

// ---------- prep kernels ----------

__global__ void k_cast_bf16(const float* __restrict__ in,
                            unsigned short* __restrict__ out, long n) {
  long i = (long)blockIdx.x * blockDim.x + threadIdx.x;
  if (i < n) out[i] = f2bf(in[i]);
}

// Wt[n][k] = bf16(W[k][n]);  W is [K][Nout] row-major
__global__ void k_transpose_cast_bf16(const float* __restrict__ W,
                                      unsigned short* __restrict__ Wt,
                                      int K, int Nout) {
  long i = (long)blockIdx.x * blockDim.x + threadIdx.x;
  if (i >= (long)K * Nout) return;
  int k = (int)(i % K);
  int n = (int)(i / K);
  Wt[(size_t)n * K + k] = f2bf(W[(size_t)k * Nout + n]);
}

__global__ void k_fill(float* __restrict__ p, float v, long n) {
  long i = (long)blockIdx.x * blockDim.x + threadIdx.x;
  if (i < n) p[i] = v;
}

// ---------- WMMA GEMM: C[M][Nc] = A[M][K](bf16) x Bt[Nc][K](bf16), fp32 acc ----------
// One wave per 16x16 C tile; 8 waves (8 M-tiles) per block; blockIdx.y = N tile.
// A fragment (16x32 bf16): lane l: row M = l%16, g = l/16.
//   v16bf elements 0..7  = K (8g .. 8g+7)       -> one 16B load
//   v16bf elements 8..15 = K (16+8g .. 16+8g+7) -> one 16B load
// B fragment (32x16 bf16): lane l: col N = l%16, same K split as A -> two 16B loads (Bt rows)
__global__ __launch_bounds__(256)
void k_wmma_gemm(const unsigned short* __restrict__ A,
                 const unsigned short* __restrict__ Bt,
                 float* __restrict__ C,
                 int M, int Nc, int K) {
  int wid  = threadIdx.x >> 5;
  int lane = threadIdx.x & 31;
  int m_tile = blockIdx.x * 8 + wid;
  int n_tile = blockIdx.y;
  if (m_tile * 16 >= M) return;

  int half = lane & 15;
  int g    = lane >> 4;
  int mrow = m_tile * 16 + half;
  if (mrow >= M) mrow = M - 1;            // clamp loads for (non-occurring) ragged tile
  int nrow = n_tile * 16 + half;

  const unsigned short* arow = A  + (size_t)mrow * K;
  const unsigned short* brow = Bt + (size_t)nrow * K;

  v8f acc = {};
  for (int k0 = 0; k0 < K; k0 += 32) {
    union { v16bf v; v8bf h[2]; } a;
    union { v16bf v; v8bf h[2]; } b;
    a.h[0] = *(const v8bf*)(arow + k0 + 8 * g);
    a.h[1] = *(const v8bf*)(arow + k0 + 16 + 8 * g);
    b.h[0] = *(const v8bf*)(brow + k0 + 8 * g);
    b.h[1] = *(const v8bf*)(brow + k0 + 16 + 8 * g);
    acc = __builtin_amdgcn_wmma_f32_16x16x32_bf16(
        false, a.v, false, b.v, (short)0, acc, false, false);
  }

  union { v8f v; float f[8]; } res;
  res.v = acc;
#pragma unroll
  for (int r = 0; r < 8; ++r) {
    int m = m_tile * 16 + r + 8 * g;      // C layout: VGPR r -> row r (+8 for upper half)
    if (m < M) C[(size_t)m * Nc + n_tile * 16 + half] = res.f[r];
  }
}

// ---------- per-node attention coefficients: alpha_s/d[n] = dot(h[n,:], a_s/d) ----------
__global__ __launch_bounds__(256)
void k_alpha(const float* __restrict__ h, const float* __restrict__ a_s,
             const float* __restrict__ a_d, float* __restrict__ as_out,
             float* __restrict__ ad_out, long N, int F) {
  int wid  = threadIdx.x >> 5;
  int lane = threadIdx.x & 31;
  long node = (long)blockIdx.x * 8 + wid;
  if (node >= N) return;
  const float* row = h + (size_t)node * F;
  float ss = 0.f, sd = 0.f;
  for (int f = lane; f < F; f += 32) {
    float v = row[f];
    ss += v * a_s[f];
    sd += v * a_d[f];
  }
#pragma unroll
  for (int off = 16; off > 0; off >>= 1) {
    ss += __shfl_down(ss, off, 32);
    sd += __shfl_down(sd, off, 32);
  }
  if (lane == 0) { as_out[node] = ss; ad_out[node] = sd; }
}

// ---------- edge pass 1: e = leakyrelu(as[src]+ad[dst]); segment max ----------
__global__ void k_edge_max(const long long* __restrict__ ei,
                           const float* __restrict__ as, const float* __restrict__ ad,
                           float* __restrict__ ebuf, float* __restrict__ mbuf,
                           long E, long Etot) {
  long i = (long)blockIdx.x * blockDim.x + threadIdx.x;
  if (i >= Etot) return;
  long s, d;
  if (i < E) { s = ei[i]; d = ei[E + i]; } else { s = d = i - E; }
  float e = as[s] + ad[d];
  e = e > 0.f ? e : NEG_ATT * e;
  ebuf[i] = e;
  atomicMaxFloat(mbuf + d, e);
}

// ---------- edge pass 2: ex = exp(e - m[dst]); segment sum ----------
__global__ void k_edge_exp(const long long* __restrict__ ei,
                           const float* __restrict__ ebuf, const float* __restrict__ mbuf,
                           float* __restrict__ exbuf, float* __restrict__ dbuf,
                           long E, long Etot) {
  long i = (long)blockIdx.x * blockDim.x + threadIdx.x;
  if (i >= Etot) return;
  long d = (i < E) ? ei[E + i] : (i - E);
  float ex = __expf(ebuf[i] - mbuf[d]);
  exbuf[i] = ex;
  atomicAdd(dbuf + d, ex);
}

// ---------- edge pass 3: agg[dst,:] += (ex/denom[dst]) * h[src,:]  (one wave/edge) ----------
__global__ __launch_bounds__(256)
void k_edge_agg(const long long* __restrict__ ei, const float* __restrict__ exbuf,
                const float* __restrict__ dbuf, const float* __restrict__ h,
                float* __restrict__ agg, long E, long Etot, int F) {
  int wid  = threadIdx.x >> 5;
  int lane = threadIdx.x & 31;
  long edge = (long)blockIdx.x * 8 + wid;
  if (edge >= Etot) return;
  long s, d;
  if (edge < E) { s = ei[edge]; d = ei[E + edge]; } else { s = d = edge - E; }
  float coef = exbuf[edge] / dbuf[d];
  const float* hs = h + (size_t)s * F;
  float* od = agg + (size_t)d * F;
  for (int f = lane; f < F; f += 32) atomicAdd(od + f, coef * hs[f]);
}

// ---------- bias + (optional) leaky-relu, recast to bf16 for next GEMM ----------
__global__ void k_post(const float* __restrict__ agg, const float* __restrict__ bias,
                       unsigned short* __restrict__ outb, long N, int F, int apply_act) {
  long i = (long)blockIdx.x * blockDim.x + threadIdx.x;
  if (i >= N * (long)F) return;
  int f = (int)(i % F);
  float v = agg[i] + bias[f];
  if (apply_act) v = v > 0.f ? v : NEG_ACT * v;
  outb[i] = f2bf(v);
}

// ---------- host ----------

extern "C" void kernel_launch(void* const* d_in, const int* in_sizes, int n_in,
                              void* d_out, int out_size, void* d_ws, size_t ws_size,
                              hipStream_t stream) {
  const float*     x   = (const float*)d_in[0];
  const long long* ei  = (const long long*)d_in[1];
  const float* W1 = (const float*)d_in[2];
  const float* a1s = (const float*)d_in[3];
  const float* a1d = (const float*)d_in[4];
  const float* b1  = (const float*)d_in[5];
  const float* W2 = (const float*)d_in[6];
  const float* a2s = (const float*)d_in[7];
  const float* a2d = (const float*)d_in[8];
  const float* b2  = (const float*)d_in[9];
  const float* W3 = (const float*)d_in[10];
  const float* a3s = (const float*)d_in[11];
  const float* a3d = (const float*)d_in[12];
  const float* b3  = (const float*)d_in[13];
  const float* S   = (const float*)d_in[14];

  const int  H1  = in_sizes[3];
  const int  H2  = in_sizes[7];
  const int  F3  = in_sizes[11];
  const int  Fin = in_sizes[2] / H1;
  const long N   = (long)in_sizes[0] / Fin;
  const long E   = (long)in_sizes[1] / 2;
  const int  Kp  = in_sizes[14] / F3;      // 512
  const long Etot = E + N;
  int Fmax = Fin;
  if (H1 > Fmax) Fmax = H1;
  if (H2 > Fmax) Fmax = H2;
  if (F3 > Fmax) Fmax = F3;

  // ---- workspace carve ----
  char* ws = (char*)d_ws;
  size_t off = 0;
  auto carve = [&](size_t bytes) -> void* {
    void* p = ws + off;
    off = (off + bytes + 255) & ~(size_t)255;
    return p;
  };
  unsigned short* bfA  = (unsigned short*)carve((size_t)N * Fmax * 2);
  float*          hbuf = (float*)carve((size_t)N * Fmax * 4);
  float*          agg  = (float*)carve((size_t)N * Fmax * 4);
  float*          asb  = (float*)carve((size_t)N * 4);
  float*          adb  = (float*)carve((size_t)N * 4);
  float*          mbuf = (float*)carve((size_t)N * 4);
  float*          dbuf = (float*)carve((size_t)N * 4);
  float*          ebuf = (float*)carve((size_t)Etot * 4);
  float*          exbuf= (float*)carve((size_t)Etot * 4);
  unsigned short* w1t  = (unsigned short*)carve((size_t)H1 * Fin * 2);
  unsigned short* w2t  = (unsigned short*)carve((size_t)H2 * H1 * 2);
  unsigned short* w3t  = (unsigned short*)carve((size_t)F3 * H2 * 2);
  unsigned short* sb   = (unsigned short*)carve((size_t)Kp * F3 * 2);
  (void)ws_size; (void)n_in; (void)out_size;

  const int TB = 256;
  auto nb = [&](long n) -> unsigned { return (unsigned)((n + TB - 1) / TB); };

  // ---- weight prep (bf16, B^T layout) ----
  k_transpose_cast_bf16<<<nb((long)Fin * H1), TB, 0, stream>>>(W1, w1t, Fin, H1);
  k_transpose_cast_bf16<<<nb((long)H1 * H2), TB, 0, stream>>>(W2, w2t, H1, H2);
  k_transpose_cast_bf16<<<nb((long)H2 * F3), TB, 0, stream>>>(W3, w3t, H2, F3);
  k_cast_bf16<<<nb((long)Kp * F3), TB, 0, stream>>>(S, sb, (long)Kp * F3);  // S is already B^T
  k_cast_bf16<<<nb(N * Fin), TB, 0, stream>>>(x, bfA, N * Fin);

  auto gemm_grid = [&](int Nc) -> dim3 {
    long m_tiles = (N + 15) / 16;
    return dim3((unsigned)((m_tiles + 7) / 8), (unsigned)(Nc / 16));
  };

  auto run_layer = [&](const unsigned short* Wt, int FinL, int FoutL,
                       const float* a_s, const float* a_d, const float* bias,
                       int apply_act) {
    // h = A_bf16 @ W  (WMMA)
    k_wmma_gemm<<<gemm_grid(FoutL), TB, 0, stream>>>(bfA, Wt, hbuf, (int)N, FoutL, FinL);
    // per-node attention terms
    k_alpha<<<(unsigned)((N + 7) / 8), TB, 0, stream>>>(hbuf, a_s, a_d, asb, adb, N, FoutL);
    // reset segment buffers
    k_fill<<<nb(N), TB, 0, stream>>>(mbuf, -3.0e38f, N);
    k_fill<<<nb(N), TB, 0, stream>>>(dbuf, 0.0f, N);
    k_fill<<<nb(N * FoutL), TB, 0, stream>>>(agg, 0.0f, N * FoutL);
    // edge softmax + aggregate
    k_edge_max<<<nb(Etot), TB, 0, stream>>>(ei, asb, adb, ebuf, mbuf, E, Etot);
    k_edge_exp<<<nb(Etot), TB, 0, stream>>>(ei, ebuf, mbuf, exbuf, dbuf, E, Etot);
    k_edge_agg<<<(unsigned)((Etot + 7) / 8), TB, 0, stream>>>(ei, exbuf, dbuf, hbuf, agg,
                                                              E, Etot, FoutL);
    // bias + activation, recast to bf16 as next layer's A
    k_post<<<nb(N * FoutL), TB, 0, stream>>>(agg, bias, bfA, N, FoutL, apply_act);
  };

  run_layer(w1t, Fin, H1, a1s, a1d, b1, 1);
  run_layer(w2t, H1, H2, a2s, a2d, b2, 1);
  run_layer(w3t, H2, F3, a3s, a3d, b3, 0);

  // out = h3 @ S^T  (WMMA, fp32 out straight to d_out)
  k_wmma_gemm<<<gemm_grid(Kp), TB, 0, stream>>>(bfA, sb, (float*)d_out, (int)N, Kp, F3);
}
